// _lstm_cell_jit_47510928229134
// MI455X (gfx1250) — compile-verified, matrix operand order checked by reading, not run on hardware
//
#include <hip/hip_runtime.h>
#include <hip/hip_bf16.h>

// ---------------- problem constants ----------------
#define Bsz   32
#define Tsz   1000
#define Hsz   512
#define G4H   2048            // 4*H
#define NWG   16              // persistent workgroups (grid size)
#define HS    32              // hidden units per WG  (H / NWG)
#define NG    128             // gate rows per WG (4*HS)
#define NTHR  256             // threads per WG (8 waves)
#define LDH   520             // padded K-row stride (bf16 elems): 260 words, %64 == 4
#define GPAD  132             // padded gates row stride (f32 elems)
#define BTH   (Bsz * Tsz * Hsz)

typedef __bf16 bf16_t;
typedef bf16_t v16bf __attribute__((ext_vector_type(16)));
typedef bf16_t v8bf  __attribute__((ext_vector_type(8)));
typedef float  v8f   __attribute__((ext_vector_type(8)));

union BF16x16 { v16bf v; v8bf h[2]; };

#if __has_builtin(__builtin_amdgcn_global_load_async_to_lds_b128)
#define HAVE_ASYNC_LDS 1
// exact parameter types per hipcc diagnostic: (v4i AS1*, v4i AS3*, imm, imm)
typedef int v4i_t __attribute__((vector_size(16)));
typedef v4i_t __attribute__((address_space(1))) g_v4i_t;
typedef v4i_t __attribute__((address_space(3))) l_v4i_t;
#endif

__device__ __forceinline__ float sigmoidf_(float x) {
    return 1.0f / (1.0f + __expf(-x));
}

// init kernel: zero per-step barrier flags, seed bf16 h state from ht
__global__ void lstm_init_kernel(const float* __restrict__ ht,
                                 bf16_t* __restrict__ h_bf,
                                 int* __restrict__ flags) {
    int i = blockIdx.x * blockDim.x + threadIdx.x;
    if (i < Tsz) flags[i] = 0;
    if (i < Bsz * Hsz) h_bf[i] = (bf16_t)ht[i];
}

__global__ __launch_bounds__(NTHR)
void lstm_scan_kernel(const float* __restrict__ wx,      // [B, T, 4H]
                      const float* __restrict__ u,       // [4H, H]
                      const float* __restrict__ u_bias,  // [4H]
                      const float* __restrict__ ct0,     // [B, H]
                      float* __restrict__ out,           // 6 x [B, T, H]
                      bf16_t* __restrict__ h_bf,         // [B, H] shared state (ws)
                      int* __restrict__ flags)           // [T] barrier flags (ws)
{
    __shared__ bf16_t u_lds[NG  * LDH];   // 128 x 520 bf16 = 130 KB
    __shared__ bf16_t h_lds[Bsz * LDH];   //  32 x 520 bf16 = 32.5 KB
    __shared__ float  g_lds[Bsz * GPAD];  //  32 x 132 f32  = 16.5 KB

    const int tid  = threadIdx.x;
    const int wg   = blockIdx.x;          // 0..15
    const int h0   = wg * HS;             // hidden-slice base
    const int lane = tid & 31;
    const int wave = tid >> 5;            // 0..7 -> N tile
    const int m    = lane & 15;           // row/col within 16-tile
    const int kh   = (lane >> 4) * 8;     // K half-select (A/B 16-bit WMMA layout)

    // ---- one-time: stage u slice into LDS as bf16 (row n -> u row gate*H + h0 + jj) ----
    for (int e = tid; e < NG * Hsz; e += NTHR) {
        int n = e >> 9, k = e & 511;
        int gate = n >> 5, jj = n & 31;
        u_lds[n * LDH + k] = (bf16_t)u[(size_t)(gate * Hsz + h0 + jj) * Hsz + k];
    }

    // ---- per-thread persistent state for the elementwise phase ----
    const int jj0 = tid & 31;             // hidden index within slice (same for all e)
    const int b0  = tid >> 5;             // batch base; b_e = b0 + 8*e
    const float bi = u_bias[0 * Hsz + h0 + jj0];
    const float bf = u_bias[1 * Hsz + h0 + jj0];
    const float bg = u_bias[2 * Hsz + h0 + jj0];
    const float bo = u_bias[3 * Hsz + h0 + jj0];
    float c_reg[4];
#pragma unroll
    for (int e = 0; e < 4; ++e)
        c_reg[e] = ct0[(size_t)(b0 + 8 * e) * Hsz + h0 + jj0];

    __syncthreads();

    for (int t = 0; t < Tsz; ++t) {
        // ---- hoisted wx loads: independent of h, latency hides under the GEMM ----
        float xw[4][4];
#pragma unroll
        for (int e = 0; e < 4; ++e) {
            const int b = b0 + 8 * e;
            const size_t wxoff = (size_t)b * (Tsz * G4H) + (size_t)t * G4H + h0 + jj0;
#pragma unroll
            for (int g = 0; g < 4; ++g)
                xw[e][g] = wx[wxoff + (size_t)g * Hsz];
            if (t + 1 < Tsz) {
                __builtin_prefetch(&wx[wxoff + G4H], 0, 1);           // global_prefetch_b8
                __builtin_prefetch(&wx[wxoff + G4H + 2 * Hsz], 0, 1);
            }
        }

        // ---- stage current h (bf16, written by all WGs last step) into LDS ----
#ifdef HAVE_ASYNC_LDS
        for (int ci = tid; ci < (Bsz * Hsz) / 8; ci += NTHR) {
            int row = ci >> 6, c8 = ci & 63;
            g_v4i_t* gsrc = (g_v4i_t*)(uintptr_t)&h_bf[row * Hsz + c8 * 8];
            l_v4i_t* ldst = (l_v4i_t*)(uintptr_t)&h_lds[row * LDH + c8 * 8];
            __builtin_amdgcn_global_load_async_to_lds_b128(gsrc, ldst, 0, 0);
        }
#if __has_builtin(__builtin_amdgcn_s_wait_asynccnt)
        __builtin_amdgcn_s_wait_asynccnt(0);
#else
        asm volatile("s_wait_asynccnt 0x0" ::: "memory");
#endif
#else
        for (int ci = tid; ci < (Bsz * Hsz) / 8; ci += NTHR) {
            int row = ci >> 6, c8 = ci & 63;
            uint4 d = *(const uint4*)&h_bf[row * Hsz + c8 * 8];
            *(uint4*)&h_lds[row * LDH + c8 * 8] = d;
        }
#endif
        __syncthreads();

        // ---- recurrent GEMM slice: gates[32 x 128] += h[32 x 512] * u_slice^T ----
        v8f acc0 = {}, acc1 = {};
#pragma unroll
        for (int kb = 0; kb < 16; ++kb) {
            const int k = kb * 32;
            BF16x16 a0, a1, bb;
            a0.h[0] = *(const v8bf*)&h_lds[m * LDH + k + kh];
            a0.h[1] = *(const v8bf*)&h_lds[m * LDH + k + 16 + kh];
            a1.h[0] = *(const v8bf*)&h_lds[(16 + m) * LDH + k + kh];
            a1.h[1] = *(const v8bf*)&h_lds[(16 + m) * LDH + k + 16 + kh];
            bb.h[0] = *(const v8bf*)&u_lds[(wave * 16 + m) * LDH + k + kh];
            bb.h[1] = *(const v8bf*)&u_lds[(wave * 16 + m) * LDH + k + 16 + kh];
            acc0 = __builtin_amdgcn_wmma_f32_16x16x32_bf16(
                       false, a0.v, false, bb.v, (short)0, acc0, false, false);
            acc1 = __builtin_amdgcn_wmma_f32_16x16x32_bf16(
                       false, a1.v, false, bb.v, (short)0, acc1, false, false);
        }
        // scatter C tiles to LDS: element v -> (row = half*8 + v, col = wave*16 + m)
        {
            const int nb = wave * 16 + m;
            const int rb = (lane >> 4) * 8;
#pragma unroll
            for (int v = 0; v < 8; ++v) {
                g_lds[(rb + v) * GPAD + nb]      = acc0[v];
                g_lds[(16 + rb + v) * GPAD + nb] = acc1[v];
            }
        }
        __syncthreads();

        // ---- elementwise LSTM cell: compute into registers, handoff h first ----
        float hv[4], cv[4], iv[4], fv[4], gv[4], ov[4];
#pragma unroll
        for (int e = 0; e < 4; ++e) {
            const int b = b0 + 8 * e;
            float ig = g_lds[b * GPAD + 0 * HS + jj0] + xw[e][0] + bi;
            float fg = g_lds[b * GPAD + 1 * HS + jj0] + xw[e][1] + bf;
            float gg = g_lds[b * GPAD + 2 * HS + jj0] + xw[e][2] + bg;
            float og = g_lds[b * GPAD + 3 * HS + jj0] + xw[e][3] + bo;
            float it = sigmoidf_(ig);
            float ft = sigmoidf_(fg);
            float gt = tanhf(gg);
            float ot = sigmoidf_(og);
            float c  = ft * c_reg[e] + it * gt;
            c_reg[e] = c;
            float h  = ot * tanhf(c);
            iv[e] = it; fv[e] = ft; gv[e] = gt; ov[e] = ot; cv[e] = c; hv[e] = h;
            h_bf[b * Hsz + h0 + jj0] = (bf16_t)h;   // handoff for next step (only store
        }                                           // that must precede the signal)

        // ---- release h handoff, signal, then drain outputs while tid0 spins ----
        __builtin_amdgcn_fence(__ATOMIC_RELEASE, "agent");
        __syncthreads();
        if (tid == 0) atomicAdd(&flags[t], 1);

#pragma unroll
        for (int e = 0; e < 4; ++e) {
            const int b = b0 + 8 * e;
            const size_t ooff = (size_t)b * (Tsz * Hsz) + (size_t)t * Hsz + h0 + jj0;
            out[0 * (size_t)BTH + ooff] = hv[e];
            out[1 * (size_t)BTH + ooff] = cv[e];
            out[2 * (size_t)BTH + ooff] = iv[e];
            out[3 * (size_t)BTH + ooff] = fv[e];
            out[4 * (size_t)BTH + ooff] = gv[e];
            out[5 * (size_t)BTH + ooff] = ov[e];
        }

        if (tid == 0) {
            while (__hip_atomic_load(&flags[t], __ATOMIC_ACQUIRE,
                                     __HIP_MEMORY_SCOPE_AGENT) < NWG) {
                __builtin_amdgcn_s_sleep(8);
            }
        }
        __syncthreads();
    }
}

extern "C" void kernel_launch(void* const* d_in, const int* in_sizes, int n_in,
                              void* d_out, int out_size, void* d_ws, size_t ws_size,
                              hipStream_t stream) {
    const float* wx     = (const float*)d_in[0];
    const float* u      = (const float*)d_in[1];
    const float* u_bias = (const float*)d_in[2];
    const float* ht     = (const float*)d_in[3];
    const float* ct     = (const float*)d_in[4];
    float* out = (float*)d_out;

    bf16_t* h_bf = (bf16_t*)d_ws;                                   // 32 KB
    int*    flags = (int*)((char*)d_ws + (size_t)Bsz * Hsz * sizeof(bf16_t));

    lstm_init_kernel<<<64, 256, 0, stream>>>(ht, h_bf, flags);
    lstm_scan_kernel<<<NWG, NTHR, 0, stream>>>(wx, u, u_bias, ct, out, h_bf, flags);
}
